// vMFMixture_46136538694124
// MI455X (gfx1250) — matrix operand order repaired and checked by reading, not compiled, not run on hardware
//
#include <hip/hip_runtime.h>
#include <hip/hip_bf16.h>
#include <math.h>

// ---------------------------------------------------------------------------
// vMF mixture: log_likelihood (N) + log_prob (N,K); N=262144, D=256, K=64
// Memory-bound (325 MB @ 23.3 TB/s ~ 14us). bf16 split-GEMM (3 WMMAs per K32
// chunk) keeps matrix math under the HBM floor with ~f32 accuracy.
// ---------------------------------------------------------------------------

typedef __attribute__((ext_vector_type(16))) __bf16 v16bf;
typedef __attribute__((ext_vector_type(16))) float  v16f;
typedef __attribute__((ext_vector_type(8)))  float  v8f;

#define D_DIM 256
#define K_COMP 64
#define NHLOG2PI (-0.91893853320467274178f)   // -0.5*log(2*pi)

__device__ __forceinline__ unsigned short f2bf(float f) {
    unsigned u = __float_as_uint(f);
    u += 0x7FFFu + ((u >> 16) & 1u);          // round-to-nearest-even
    return (unsigned short)(u >> 16);
}
__device__ __forceinline__ float bf2f(unsigned short h) {
    return __uint_as_float(((unsigned)h) << 16);
}

union BF16x16 {
    unsigned short us[16];
    uint4          u4[2];
    v16bf          v;
};
union F32x16 {
    float  f[16];
    float4 f4[4];
    v16f   v;
};

// ---------------------------------------------------------------------------
// Kernel 1: per-component params + B-matrix packing (bf16 hi/lo split),
// stored in a contiguous-k B-operand lane layout (same k order as A in vmf_main):
//   lane<16  : col=lane,    K = 0..15  (elements 0..15)
//   lane>=16 : col=lane-16, K = 16..31 (elements 0..15)
// ---------------------------------------------------------------------------
__global__ void vmf_prep(const float* __restrict__ alpha_logit,
                         const float* __restrict__ mu_unnorm,
                         const float* __restrict__ log_kappa,
                         unsigned short* __restrict__ Bhi,
                         unsigned short* __restrict__ Blo,
                         float* __restrict__ pKappa,
                         float* __restrict__ pLogC,
                         float* __restrict__ pLogAlpha)
{
    __shared__ float red[256];
    const int k = blockIdx.x;       // component
    const int d = threadIdx.x;      // dimension

    float val = mu_unnorm[k * D_DIM + d];
    red[d] = val * val;
    __syncthreads();
    for (int off = 128; off > 0; off >>= 1) {
        if (d < off) red[d] += red[d + off];
        __syncthreads();
    }
    float norm = fmaxf(sqrtf(red[0]), 1e-12f);
    float mu = val / norm;

    unsigned short hi = f2bf(mu);
    unsigned short lo = f2bf(mu - bf2f(hi));

    const int kb   = d >> 5;          // which K-32 chunk (0..7)
    const int Kloc = d & 31;          // K within chunk
    const int cg   = k >> 4;          // comp group (0..3)
    const int col  = k & 15;          // column within 16-wide tile
    const int lane = ((Kloc >> 4) << 4) | col;
    const int e    = Kloc & 15;
    const int idx  = (((kb * 4 + cg) * 32) + lane) * 16 + e;
    Bhi[idx] = hi;
    Blo[idx] = lo;

    if (d == 0) {
        // kappa, logC via stable streaming logsumexp of the Bessel series
        float kap = expf(log_kappa[k]) + 1e-6f;
        float s   = 0.5f * (float)D_DIM - 1.0f;      // 127
        float lk2 = logf(0.5f * kap);
        float mrun = -INFINITY, srun = 0.0f;
        for (int m = 0; m < 64; ++m) {
            float fm = (float)m;
            float t = (2.0f * fm + s) * lk2 - lgammaf(fm + 1.0f) - lgammaf(fm + s + 1.0f);
            if (t > mrun) { srun = srun * expf(mrun - t) + 1.0f; mrun = t; }
            else          { srun += expf(t - mrun); }
        }
        float logI = mrun + logf(srun);
        pKappa[k] = kap;
        pLogC[k]  = (float)D_DIM * NHLOG2PI + s * logf(kap) - logI;

        // log_softmax over alpha_logit (K=64, trivial)
        float amax = -INFINITY;
        for (int j = 0; j < K_COMP; ++j) amax = fmaxf(amax, alpha_logit[j]);
        float se = 0.0f;
        for (int j = 0; j < K_COMP; ++j) se += expf(alpha_logit[j] - amax);
        pLogAlpha[k] = alpha_logit[k] - (amax + logf(se));
    }
}

// ---------------------------------------------------------------------------
// Kernel 2: streaming GEMM + fused epilogue.
// Each wave: 16 rows x 64 comps via 4 f32 accumulators.
// A-chunks double-buffered (unroll 2 so buffers alternate without copies);
// next grid-stride tile prefetched with global_prefetch during compute.
// ---------------------------------------------------------------------------
__global__ void __launch_bounds__(256) vmf_main(
    const float* __restrict__ x,
    const unsigned short* __restrict__ Bhi,
    const unsigned short* __restrict__ Blo,
    const float* __restrict__ pKappa,
    const float* __restrict__ pLogC,
    const float* __restrict__ pLogAlpha,
    float* __restrict__ out_ll,
    float* __restrict__ out_lp,
    int N)
{
    // packed B tiles: 8 kb * 4 cg * 32 lanes * 16 elems = 16384 bf16 = 32 KB each
    __shared__ __attribute__((aligned(16))) unsigned short sBhi[8 * 4 * 32 * 16];
    __shared__ __attribute__((aligned(16))) unsigned short sBlo[8 * 4 * 32 * 16];
    {
        const uint4* gh = (const uint4*)Bhi;
        const uint4* gl = (const uint4*)Blo;
        uint4* sh = (uint4*)sBhi;
        uint4* sl = (uint4*)sBlo;
        for (int i = threadIdx.x; i < 2048; i += 256) { sh[i] = gh[i]; sl[i] = gl[i]; }
    }
    __syncthreads();

    const int lane = threadIdx.x & 31;
    const int wave = threadIdx.x >> 5;
    const int col  = lane & 15;     // N-column within 16-wide tile (also A row sel)
    const int h    = lane >> 4;     // half-wave selector

    // per-lane component constants (comp = g*16 + col), invariant across tiles
    float kap[4], lC[4], lA[4];
#pragma unroll
    for (int g = 0; g < 4; ++g) {
        int c = g * 16 + col;
        kap[g] = pKappa[c]; lC[g] = pLogC[c]; lA[g] = pLogAlpha[c];
    }

    // per-lane LDS bases (lane-contiguous 32B rows)
    const unsigned short* sBhiL = sBhi + lane * 16;
    const unsigned short* sBloL = sBlo + lane * 16;

    const int tiles = (N + 15) >> 4;
    const int totalWaves = gridDim.x * 8;
    for (int tile = blockIdx.x * 8 + wave; tile < tiles; tile += totalWaves) {
        const int m0 = tile << 4;

        // A-operand row for this lane (contiguous-k order matching packed B, 16x32):
        //   lane<16 : row=lane,    holds K 0..15
        //   lane>=16: row=lane-16, holds K 16..31
        int r = m0 + col;
        if (r >= N) r = N - 1;
        const float* xrow = x + (size_t)r * D_DIM + h * 16;

        // warm L2 for the next grid-stride tile while this one computes
        {
            const int tn = tile + totalWaves;
            if (tn < tiles) {
                const float* pf = x + (size_t)((tn << 4) + col) * D_DIM + h * 16;
#pragma unroll
                for (int kb = 0; kb < 8; ++kb)
                    __builtin_prefetch(pf + kb * 32, 0, 0);
            }
        }

        v8f acc[4] = {};

        // prime A double-buffer with chunk kb=0
        F32x16 a;
        a.f4[0] = *(const float4*)(xrow);
        a.f4[1] = *(const float4*)(xrow + 4);
        a.f4[2] = *(const float4*)(xrow + 8);
        a.f4[3] = *(const float4*)(xrow + 12);

#pragma unroll 2
        for (int kb = 0; kb < 8; ++kb) {
            // issue next chunk's loads early ((kb+1)&7 wraps to an in-cache chunk)
            const float* xn = xrow + (((kb + 1) & 7) * 32);
            F32x16 an;
            an.f4[0] = *(const float4*)(xn);
            an.f4[1] = *(const float4*)(xn + 4);
            an.f4[2] = *(const float4*)(xn + 8);
            an.f4[3] = *(const float4*)(xn + 12);

            // bf16 hi/lo split with native packed converts (RNE)
            BF16x16 ahi, alo;
            ahi.v = __builtin_convertvector(a.v, v16bf);
            v16f hif = __builtin_convertvector(ahi.v, v16f);
            alo.v = __builtin_convertvector(a.v - hif, v16bf);

#pragma unroll
            for (int cg = 0; cg < 4; ++cg) {
                const int tb = (kb * 4 + cg) * 32 * 16;   // tile base (elements)
                BF16x16 bhi, blo;
                const uint4* ph = (const uint4*)(sBhiL + tb);
                const uint4* pl = (const uint4*)(sBloL + tb);
                bhi.u4[0] = ph[0]; bhi.u4[1] = ph[1];
                blo.u4[0] = pl[0]; blo.u4[1] = pl[1];
                // split product: hi*hi + hi*lo + lo*hi (lo*lo below f32 ulp)
                acc[cg] = __builtin_amdgcn_wmma_f32_16x16x32_bf16(
                    false, ahi.v, false, bhi.v, (short)0, acc[cg], false, false);
                acc[cg] = __builtin_amdgcn_wmma_f32_16x16x32_bf16(
                    false, ahi.v, false, blo.v, (short)0, acc[cg], false, false);
                acc[cg] = __builtin_amdgcn_wmma_f32_16x16x32_bf16(
                    false, alo.v, false, bhi.v, (short)0, acc[cg], false, false);
            }
            a = an;
        }

        // Fused epilogue. C/D layout: lane<16 -> (M=v, N=lane); lane>=16 -> (M=8+v, N=lane-16)
        const bool full = (m0 + 16 <= N);   // wave-uniform: branch-free hot path
#pragma unroll
        for (int v = 0; v < 8; ++v) {
            const int m = m0 + v + (h << 3);
            float lp[4], t[4];
            float mx = -INFINITY;
#pragma unroll
            for (int g = 0; g < 4; ++g) {
                lp[g] = fmaf(kap[g], acc[g][v], lC[g]);
                t[g]  = lp[g] + lA[g];
                mx = fmaxf(mx, t[g]);
            }
            if (full) {
#pragma unroll
                for (int g = 0; g < 4; ++g)
                    out_lp[(size_t)m * K_COMP + g * 16 + col] = lp[g];
            } else if (m < N) {
#pragma unroll
                for (int g = 0; g < 4; ++g)
                    out_lp[(size_t)m * K_COMP + g * 16 + col] = lp[g];
            }
            // cross-lane max within the 16-lane half (xor 1/2/4/8 stays in-half)
#pragma unroll
            for (int off = 1; off < 16; off <<= 1)
                mx = fmaxf(mx, __shfl_xor(mx, off, 32));
            float se = 0.0f;
#pragma unroll
            for (int g = 0; g < 4; ++g) se += expf(t[g] - mx);
#pragma unroll
            for (int off = 1; off < 16; off <<= 1)
                se += __shfl_xor(se, off, 32);
            if (col == 0 && m < N) out_ll[m] = mx + logf(se);
        }
    }
}

// ---------------------------------------------------------------------------
extern "C" void kernel_launch(void* const* d_in, const int* in_sizes, int n_in,
                              void* d_out, int out_size, void* d_ws, size_t ws_size,
                              hipStream_t stream) {
    const float* x           = (const float*)d_in[0];
    const float* alpha_logit = (const float*)d_in[1];
    const float* mu_unnorm   = (const float*)d_in[2];
    const float* log_kappa   = (const float*)d_in[3];

    const int N = in_sizes[0] / D_DIM;

    // workspace layout
    char* ws = (char*)d_ws;
    unsigned short* Bhi = (unsigned short*)(ws);            // 32768 B
    unsigned short* Blo = (unsigned short*)(ws + 32768);    // 32768 B
    float* pKappa = (float*)(ws + 65536);                   // 256 B
    float* pLogC  = (float*)(ws + 65536 + 256);             // 256 B
    float* pLogA  = (float*)(ws + 65536 + 512);             // 256 B

    float* out    = (float*)d_out;
    float* out_ll = out;            // (N,)
    float* out_lp = out + N;        // (N, K)

    vmf_prep<<<K_COMP, 256, 0, stream>>>(alpha_logit, mu_unnorm, log_kappa,
                                         Bhi, Blo, pKappa, pLogC, pLogA);
    vmf_main<<<512, 256, 0, stream>>>(x, Bhi, Blo, pKappa, pLogC, pLogA,
                                      out_ll, out_lp, N);
}
